// LightAttentionClassifier_80307298501414
// MI455X (gfx1250) — matrix operand order repaired and checked by reading, not compiled, not run on hardware
//
#include <hip/hip_runtime.h>
#include <hip/hip_bf16.h>

// ---------------------------------------------------------------------------
// LightAttentionClassifier on gfx1250 (MI455X).
// Convs (99.9% of FLOPs) run on v_wmma_f32_16x16x32_bf16. The transposed
// activation tensor carries a 4-row zero halo so the 'same'-padding taps need
// no predication in the WMMA inner loop (pure loads + WMMA).
// Memory-bound tail layers run as exact f32 scalar GEMMs.
// ---------------------------------------------------------------------------

typedef __attribute__((ext_vector_type(16))) __bf16        v16bf;
typedef __attribute__((ext_vector_type(8)))  float         v8f;
typedef __attribute__((ext_vector_type(8)))  unsigned short u8s;

union F16x16 { v16bf v; u8s h[2]; };

#define B_SZ   32
#define D_SZ   1280
#define L_SZ   1024
#define LPAD   (L_SZ + 8)     // 4-row zero halo each side
#define KTAPS  9
#define TWO_D  2560
#define SE_SZ  640
#define EPSV   1e-5f

__device__ __forceinline__ unsigned short f2bf(float f) {
  unsigned int u = __float_as_uint(f);
  u += 0x7fffu + ((u >> 16) & 1u);   // round-to-nearest-even
  return (unsigned short)(u >> 16);
}

// --- zero the 4-row halos of xt[B][LPAD][D] --------------------------------
__global__ __launch_bounds__(256) void zero_halo(unsigned short* __restrict__ xt) {
  int idx = blockIdx.x * 256 + threadIdx.x;          // over B*8*D
  if (idx >= B_SZ * 8 * D_SZ) return;
  int d  = idx % D_SZ;
  int r8 = (idx / D_SZ) % 8;
  int b  = idx / (D_SZ * 8);
  int row = (r8 < 4) ? r8 : (L_SZ + r8);             // 0..3 and L+4..L+7
  xt[((size_t)b * LPAD + row) * D_SZ + d] = 0;
}

// --- x[B][D][L] f32  ->  xt[B][LPAD][D] bf16 (tiled transpose, +4 row off) -
__global__ __launch_bounds__(256) void prep_x(const float* __restrict__ x,
                                              unsigned short* __restrict__ xt) {
  __shared__ float tile[32][33];
  const int b  = blockIdx.z;
  const int d0 = blockIdx.y * 32;
  const int l0 = blockIdx.x * 32;
  const int tx = threadIdx.x;           // 0..31
  for (int i = threadIdx.y; i < 32; i += 8)
    tile[i][tx] = x[((size_t)b * D_SZ + d0 + i) * L_SZ + l0 + tx];
  __syncthreads();
  for (int i = threadIdx.y; i < 32; i += 8)
    xt[((size_t)b * LPAD + l0 + i + 4) * D_SZ + d0 + tx] = f2bf(tile[tx][i]);
}

// --- W[D][D][9] f32 (x2 convs) -> wb[conv][k][dout][din] bf16 --------------
__global__ __launch_bounds__(256) void prep_w(const float* __restrict__ wf,
                                              const float* __restrict__ wa,
                                              unsigned short* __restrict__ wb) {
  size_t idx = (size_t)blockIdx.x * 256 + threadIdx.x;
  if (idx >= (size_t)D_SZ * D_SZ) return;
  const int din  = (int)(idx % D_SZ);
  const int dout = (int)(idx / D_SZ);
  const float* pf = wf + ((size_t)dout * D_SZ + din) * KTAPS;
  const float* pa = wa + ((size_t)dout * D_SZ + din) * KTAPS;
  #pragma unroll
  for (int k = 0; k < KTAPS; ++k) {
    wb[((size_t)k * D_SZ + dout) * D_SZ + din]           = f2bf(pf[k]);
    wb[((size_t)(KTAPS + k) * D_SZ + dout) * D_SZ + din] = f2bf(pa[k]);
  }
}

// --- fused fconv+aconv via WMMA bf16 ---------------------------------------
// block = 256 (8 waves). Block tile: 1 batch x 64 douts x 64 l's.
// wave (wm = wave&3, wn = 2*(wave>>2)) owns M-subtile wm, N-subtiles wn,wn+1,
// for BOTH convs (shares the B fragments). Halo padding -> zero predication.
__global__ __launch_bounds__(256) void conv2_wmma(
    const unsigned short* __restrict__ xt,   // [B][LPAD][D] bf16, halo'd
    const unsigned short* __restrict__ wb,   // [2][9][D][D] bf16
    const float* __restrict__ fbias, const float* __restrict__ abias,
    float* __restrict__ O, float* __restrict__ A) {
  const int lane = threadIdx.x & 31;
  const int wave = threadIdx.x >> 5;
  const int wm   = wave & 3;
  const int wn   = (wave >> 2) << 1;
  const int hi   = lane >> 4;
  const int ln15 = lane & 15;

  const int b  = blockIdx.z;
  const int m0 = blockIdx.y * 64;
  const int l0 = blockIdx.x * 64;
  const int mrow = m0 + wm * 16 + ln15;

  v8f accF0 = {0,0,0,0,0,0,0,0};
  v8f accF1 = accF0, accA0 = accF0, accA1 = accF0;

  const unsigned short* xb = xt + (size_t)b * LPAD * D_SZ;
  const int koffA = hi ? 16 : 0;   // A: lane<16 holds K 0..15; hi: 16..31
  const int koffB = hi ? 16 : 0;   // B: lane<16 holds K 0..15; hi: 16..31

  for (int k = 0; k < KTAPS; ++k) {
    // padded row index: x[l + k - 4] == xt_pad[l + k], always in [0, LPAD)
    const int la = l0 + wn * 16 + ln15 + k;
    const unsigned short* xra = xb + (size_t)la * D_SZ + koffB;
    const unsigned short* xrb = xra + (size_t)16 * D_SZ;
    const unsigned short* wfr = wb + ((size_t)k * D_SZ + mrow) * D_SZ + koffA;
    const unsigned short* war = wfr + (size_t)KTAPS * D_SZ * D_SZ;

    for (int d0 = 0; d0 < D_SZ; d0 += 32) {
      F16x16 af, aa, b0, b1;
      af.h[0] = *(const u8s*)(wfr + d0);
      af.h[1] = *(const u8s*)(wfr + d0 + 8);
      aa.h[0] = *(const u8s*)(war + d0);
      aa.h[1] = *(const u8s*)(war + d0 + 8);
      b0.h[0] = *(const u8s*)(xra + d0);
      b0.h[1] = *(const u8s*)(xra + d0 + 8);
      b1.h[0] = *(const u8s*)(xrb + d0);
      b1.h[1] = *(const u8s*)(xrb + d0 + 8);

      accF0 = __builtin_amdgcn_wmma_f32_16x16x32_bf16(false, af.v, false, b0.v, (short)0, accF0, false, false);
      accF1 = __builtin_amdgcn_wmma_f32_16x16x32_bf16(false, af.v, false, b1.v, (short)0, accF1, false, false);
      accA0 = __builtin_amdgcn_wmma_f32_16x16x32_bf16(false, aa.v, false, b0.v, (short)0, accA0, false, false);
      accA1 = __builtin_amdgcn_wmma_f32_16x16x32_bf16(false, aa.v, false, b1.v, (short)0, accA1, false, false);
    }
  }

  const int lc = l0 + wn * 16 + ln15;
  #pragma unroll
  for (int r = 0; r < 8; ++r) {
    const int m = m0 + wm * 16 + (hi ? r + 8 : r);
    const size_t base = ((size_t)b * D_SZ + m) * L_SZ + lc;
    O[base]      = accF0[r] + fbias[m];
    O[base + 16] = accF1[r] + fbias[m];
    A[base]      = accA0[r] + abias[m];
    A[base + 16] = accA1[r] + abias[m];
  }
}

// --- masked softmax attention pooling + unmasked max -----------------------
__global__ __launch_bounds__(32) void pool_attn(
    const float* __restrict__ O, const float* __restrict__ A,
    const unsigned char* __restrict__ mask, float* __restrict__ pooled) {
  const int d = blockIdx.x, b = blockIdx.y, lane = threadIdx.x;
  const float* orow = O + ((size_t)b * D_SZ + d) * L_SZ;
  const float* arow = A + ((size_t)b * D_SZ + d) * L_SZ;
  const unsigned char* mrow = mask + (size_t)b * L_SZ;

  float amax = -3.4e38f, omax = -3.4e38f;
  for (int l = lane; l < L_SZ; l += 32) {
    float av = mrow[l] ? arow[l] : -1e9f;
    amax = fmaxf(amax, av);
    omax = fmaxf(omax, orow[l]);
  }
  for (int off = 16; off; off >>= 1) {
    amax = fmaxf(amax, __shfl_xor(amax, off));
    omax = fmaxf(omax, __shfl_xor(omax, off));
  }
  float s = 0.f, ws = 0.f;
  for (int l = lane; l < L_SZ; l += 32) {
    float av = mrow[l] ? arow[l] : -1e9f;
    float e = __expf(av - amax);
    s += e;  ws += e * orow[l];
  }
  for (int off = 16; off; off >>= 1) {
    s  += __shfl_xor(s, off);
    ws += __shfl_xor(ws, off);
  }
  if (lane == 0) {
    pooled[(size_t)b * TWO_D + d]        = ws / s;
    pooled[(size_t)b * TWO_D + D_SZ + d] = omax;
  }
}

// --- row-wave GEMM: lane == batch (B==32==wave32!), weight addr is uniform -> SMEM
// mode 0: out = acc + bias
// mode 1: aw = sigmoid(acc+bias); out = extra * ((1-g) + g*aw)
// mode 2: out = relu((acc+bias - bn_m)*rsqrt(bn_v+eps)*bn_g + bn_b)
__global__ __launch_bounds__(32) void rowgemm(
    const float* __restrict__ Z, const float* __restrict__ W,
    const float* __restrict__ bias, float* __restrict__ Out,
    int K, int J, int mode,
    const float* __restrict__ bn_g, const float* __restrict__ bn_b,
    const float* __restrict__ bn_m, const float* __restrict__ bn_v,
    const float* __restrict__ extra, const float* __restrict__ gatep) {
  const int j = blockIdx.x;
  const int b = threadIdx.x;          // lane = batch
  const float* zr = Z + (size_t)b * K;
  float acc = 0.f;
  #pragma unroll 4
  for (int k = 0; k < K; ++k) acc = fmaf(zr[k], W[(size_t)k * J + j], acc);
  acc += bias[j];
  if (mode == 1) {
    const float g = *gatep;
    const float aw = 1.f / (1.f + __expf(-acc));
    acc = extra[(size_t)b * J + j] * ((1.f - g) + g * aw);
  } else if (mode == 2) {
    acc = (acc - bn_m[j]) * rsqrtf(bn_v[j] + EPSV) * bn_g[j] + bn_b[j];
    acc = acc > 0.f ? acc : 0.f;
  }
  Out[(size_t)b * J + j] = acc;
}

// --- LayerNorm + ReLU over last dim, in place ------------------------------
__global__ __launch_bounds__(256) void ln_relu(float* __restrict__ h,
                                               const float* __restrict__ g,
                                               const float* __restrict__ beta, int n) {
  const int b = blockIdx.x;
  float* row = h + (size_t)b * n;
  __shared__ float ss[8], ss2[8], stat[2];
  float s = 0.f, s2 = 0.f;
  for (int i = threadIdx.x; i < n; i += blockDim.x) { float v = row[i]; s += v; s2 += v * v; }
  for (int off = 16; off; off >>= 1) { s += __shfl_xor(s, off); s2 += __shfl_xor(s2, off); }
  if ((threadIdx.x & 31) == 0) { ss[threadIdx.x >> 5] = s; ss2[threadIdx.x >> 5] = s2; }
  __syncthreads();
  if (threadIdx.x == 0) {
    float a = 0.f, c = 0.f;
    for (int i = 0; i < 8; ++i) { a += ss[i]; c += ss2[i]; }
    const float mu = a / n;
    stat[0] = mu;
    stat[1] = c / n - mu * mu;
  }
  __syncthreads();
  const float mu = stat[0];
  const float inv = rsqrtf(stat[1] + EPSV);
  for (int i = threadIdx.x; i < n; i += blockDim.x) {
    float v = (row[i] - mu) * inv * g[i] + beta[i];
    row[i] = v > 0.f ? v : 0.f;
  }
}

// ---------------------------------------------------------------------------
extern "C" void kernel_launch(void* const* d_in, const int* in_sizes, int n_in,
                              void* d_out, int out_size, void* d_ws, size_t ws_size,
                              hipStream_t stream) {
  const float* x        = (const float*)d_in[0];
  const unsigned char* mask = (const unsigned char*)d_in[1];   // jax bool = 1 byte
  const float* fconv_w  = (const float*)d_in[2];
  const float* fconv_b  = (const float*)d_in[3];
  const float* aconv_w  = (const float*)d_in[4];
  const float* aconv_b  = (const float*)d_in[5];
  const float* se_w1    = (const float*)d_in[6];
  const float* se_b1    = (const float*)d_in[7];
  const float* ln_g     = (const float*)d_in[8];
  const float* ln_b     = (const float*)d_in[9];
  const float* se_w2    = (const float*)d_in[10];
  const float* se_b2    = (const float*)d_in[11];
  const float* gate     = (const float*)d_in[12];
  const float* head_w   = (const float*)d_in[13];
  const float* head_b   = (const float*)d_in[14];
  const float *mw[5], *mb[5], *mg[5], *mbb[5], *mm[5], *mv[5];
  for (int i = 0; i < 5; ++i) {
    int base = 15 + 6 * i;
    mw[i]  = (const float*)d_in[base + 0];
    mb[i]  = (const float*)d_in[base + 1];
    mg[i]  = (const float*)d_in[base + 2];
    mbb[i] = (const float*)d_in[base + 3];
    mm[i]  = (const float*)d_in[base + 4];
    mv[i]  = (const float*)d_in[base + 5];
  }

  // workspace carve-up (all offsets 256B aligned)
  char* ws = (char*)d_ws;
  size_t off = 0;
  unsigned short* xt  = (unsigned short*)(ws + off); off += (size_t)B_SZ * LPAD * D_SZ * 2;      // 84.5 MB
  unsigned short* wbq = (unsigned short*)(ws + off); off += (size_t)2 * KTAPS * D_SZ * D_SZ * 2; // 59.0 MB
  float* Obuf   = (float*)(ws + off); off += (size_t)B_SZ * D_SZ * L_SZ * 4;                     // 167.8 MB
  float* Abuf   = (float*)(ws + off); off += (size_t)B_SZ * D_SZ * L_SZ * 4;                     // 167.8 MB
  float* pooled = (float*)(ws + off); off += (size_t)B_SZ * TWO_D * 4;
  float* hbuf   = (float*)(ws + off); off += (size_t)B_SZ * SE_SZ * 4;
  float* z0     = (float*)(ws + off); off += (size_t)B_SZ * TWO_D * 4;
  float* z1     = (float*)(ws + off); off += (size_t)B_SZ * TWO_D * 4;
  (void)ws_size; (void)n_in; (void)in_sizes; (void)out_size;

  // 1) layout/precision prep (halo zero + transpose + weight repack)
  zero_halo<<<(B_SZ * 8 * D_SZ + 255) / 256, 256, 0, stream>>>(xt);
  prep_x<<<dim3(L_SZ / 32, D_SZ / 32, B_SZ), dim3(32, 8), 0, stream>>>(x, xt);
  prep_w<<<(D_SZ * D_SZ + 255) / 256, 256, 0, stream>>>(fconv_w, aconv_w, wbq);

  // 2) fused convs on WMMA
  conv2_wmma<<<dim3(L_SZ / 64, D_SZ / 64, B_SZ), 256, 0, stream>>>(
      xt, wbq, fconv_b, aconv_b, Obuf, Abuf);

  // 3) masked-softmax attention pool + max pool
  pool_attn<<<dim3(D_SZ, B_SZ), 32, 0, stream>>>(Obuf, Abuf, mask, pooled);

  // 4) SE block
  rowgemm<<<SE_SZ, 32, 0, stream>>>(pooled, se_w1, se_b1, hbuf, TWO_D, SE_SZ, 0,
                                    nullptr, nullptr, nullptr, nullptr, nullptr, nullptr);
  ln_relu<<<B_SZ, 256, 0, stream>>>(hbuf, ln_g, ln_b, SE_SZ);
  rowgemm<<<TWO_D, 32, 0, stream>>>(hbuf, se_w2, se_b2, z0, SE_SZ, TWO_D, 1,
                                    nullptr, nullptr, nullptr, nullptr, pooled, gate);

  // 5) MLP stack: Linear + BN(inference) + ReLU
  const int hid[5] = {2560, 1280, 640, 320, 160};
  float* bufs[2] = {z0, z1};
  int prev = TWO_D, cur = 0;
  for (int i = 0; i < 5; ++i) {
    rowgemm<<<hid[i], 32, 0, stream>>>(bufs[cur], mw[i], mb[i], bufs[cur ^ 1],
                                       prev, hid[i], 2,
                                       mg[i], mbb[i], mm[i], mv[i], nullptr, nullptr);
    prev = hid[i];
    cur ^= 1;
  }

  // 6) classifier head -> d_out [32, 10] f32
  rowgemm<<<10, 32, 0, stream>>>(bufs[cur], head_w, head_b, (float*)d_out, 160, 10, 0,
                                 nullptr, nullptr, nullptr, nullptr, nullptr, nullptr);
}
